// custom2D_27255862460876
// MI455X (gfx1250) — compile-verified, matrix operand order checked by reading, not run on hardware
//
#include <hip/hip_runtime.h>

typedef __attribute__((ext_vector_type(16))) _Float16 v16h;
typedef __attribute__((ext_vector_type(8)))  float    v8f;

#define LETTERS_PER_WAVE 8
#define WAVES_PER_BLOCK  8
#define THREADS_PER_BLOCK (WAVES_PER_BLOCK * 32)

// One wave = 32 lanes, wave32. One v_wmma_f32_16x16x32_f16 computes the full
// 3x3 VALID conv of one 16x8 letter:
//   D[m][n] = sum_{r=0..2, i=0..7} A[m][8r+i] * B[8r+i][n]
//   A[m][8r+i] = letter[m+r][i]   (0 if m+r > 15)
//   B[8r+i][n] = kernel[r][i-n] if 0<=i-n<=2 and n<6, else 0
// D[m][n] (m<14, n<6) == out[m*6+n].
__global__ __launch_bounds__(THREADS_PER_BLOCK)
void conv3x3_letters_wmma(const float* __restrict__ x,
                          const float* __restrict__ kern,
                          float* __restrict__ out,
                          int nLetters)
{
    const int lane = threadIdx.x & 31;
    const int wave = blockIdx.x * WAVES_PER_BLOCK + (threadIdx.x >> 5);
    const int n    = lane & 15;   // B/D column, also A row id m
    const int hi   = lane >> 4;   // lane half

    // ---- load 3x3 kernel (wave-uniform, 9 floats) ----
    float k[9];
#pragma unroll
    for (int i = 0; i < 9; ++i) k[i] = kern[i];

    // ---- build Toeplitz B (32x16 f16), column n per lane ----
    // 16-bit operand layout: lanes 0-15 hold K={0..7,16..23}, lanes 16-31
    // hold K={8..15,24..31} (elements 0..7 then 8..15).
    v16h b;
#pragma unroll
    for (int e = 0; e < 16; ++e) {
        const int K = e + 8 * hi + (e & 8);   // e<8 -> K=e+8*hi ; e>=8 -> +8 more
        const int r = K >> 3;
        const int i = K & 7;
        const int d = i - n;
        float v = 0.0f;
        if (K < 24 && n < 6 && d >= 0 && d <= 2) v = k[r * 3 + d];
        b[e] = (_Float16)v;
    }

    // A row sources for this lane:
    //   lanes 0-15 : elems 0..7 = letter row m   (r=0), elems 8..15 = row m+2 (r=2)
    //   lanes 16-31: elems 0..7 = letter row m+1 (r=1), elems 8..15 = zero pad
    const int m      = n;
    const int rowLo  = hi ? (m + 1) : m;        // rows 0..16 (16 => zero)
    const int rowHi  = hi ? 16      : (m + 2);  // rows 2..17 / 16 (>=16 => zero)

    const long base = (long)wave * LETTERS_PER_WAVE;

    for (int l = 0; l < LETTERS_PER_WAVE; ++l) {
        const long letter = base + l;
        if (letter >= (long)nLetters) break;          // wave-uniform exit
        const float* __restrict__ px = x + letter * 128;

        // prefetch the next letter this wave will touch (global_prefetch_b8)
        __builtin_prefetch(px + 128, 0, 1);

        float4 lo0 = {0.f, 0.f, 0.f, 0.f}, lo1 = {0.f, 0.f, 0.f, 0.f};
        float4 hi0 = {0.f, 0.f, 0.f, 0.f}, hi1 = {0.f, 0.f, 0.f, 0.f};
        if (rowLo < 16) {                              // predicated 128-bit loads
            lo0 = *(const float4*)(px + rowLo * 8);
            lo1 = *(const float4*)(px + rowLo * 8 + 4);
        }
        if (rowHi < 16) {
            hi0 = *(const float4*)(px + rowHi * 8);
            hi1 = *(const float4*)(px + rowHi * 8 + 4);
        }

        v16h a;
        a[0]  = (_Float16)lo0.x;  a[1]  = (_Float16)lo0.y;
        a[2]  = (_Float16)lo0.z;  a[3]  = (_Float16)lo0.w;
        a[4]  = (_Float16)lo1.x;  a[5]  = (_Float16)lo1.y;
        a[6]  = (_Float16)lo1.z;  a[7]  = (_Float16)lo1.w;
        a[8]  = (_Float16)hi0.x;  a[9]  = (_Float16)hi0.y;
        a[10] = (_Float16)hi0.z;  a[11] = (_Float16)hi0.w;
        a[12] = (_Float16)hi1.x;  a[13] = (_Float16)hi1.y;
        a[14] = (_Float16)hi1.z;  a[15] = (_Float16)hi1.w;

        v8f c = {0.f, 0.f, 0.f, 0.f, 0.f, 0.f, 0.f, 0.f};
        // EXEC is all-ones here (divergent load region closed above).
        c = __builtin_amdgcn_wmma_f32_16x16x32_f16(
                /*neg_a=*/false, a, /*neg_b=*/false, b,
                /*c_mod=*/(short)0, c, /*reuse_a=*/false, /*reuse_b=*/false);

        // D layout: element e of lane L -> row M = e + 8*(L>=16), col N = L&15.
        float* __restrict__ po = out + letter * 84;
        if (n < 6) {
#pragma unroll
            for (int e = 0; e < 8; ++e) {
                const int M = e + 8 * hi;
                if (M < 14) po[M * 6 + n] = c[e];
            }
        }
    }
}

extern "C" void kernel_launch(void* const* d_in, const int* in_sizes, int n_in,
                              void* d_out, int out_size, void* d_ws, size_t ws_size,
                              hipStream_t stream) {
    const float* x    = (const float*)d_in[0];   // [32768, 14, 128] f32
    const float* kern = (const float*)d_in[1];   // [3, 3] f32
    float* out        = (float*)d_out;           // [1, 32768, 14, 84] f32

    const int nLetters = in_sizes[0] / 128;      // 458,752
    const int lettersPerBlock = WAVES_PER_BLOCK * LETTERS_PER_WAVE;
    const int blocks = (nLetters + lettersPerBlock - 1) / lettersPerBlock;

    hipLaunchKernelGGL(conv3x3_letters_wmma, dim3(blocks), dim3(THREADS_PER_BLOCK),
                       0, stream, x, kern, out, nLetters);
}
